// CTCModel_38208029065884
// MI455X (gfx1250) — matrix operand order repaired and checked by reading, not compiled, not run on hardware
//
#include <hip/hip_runtime.h>
#include <hip/hip_bf16.h>

typedef __attribute__((ext_vector_type(16))) __bf16 v16bf;
typedef __attribute__((ext_vector_type(8)))  __bf16 v8bf;
typedef __attribute__((ext_vector_type(8)))  float  v8f;

#define B_ 32
#define T_ 1000
#define C_ 512
#define V_ 2000
#define VP_ 2048          // V padded to 16*128 column tiles
#define U_ 100
#define S_ 201
#define NEGF (-1.0e30f)
#define LP_BYTES ((size_t)B_ * T_ * S_ * 4)   // 25,728,000 (16B aligned)

static __device__ __forceinline__ v16bf cat8(v8bf lo, v8bf hi) {
  return __builtin_shufflevector(lo, hi, 0,1,2,3,4,5,6,7,8,9,10,11,12,13,14,15);
}

// ---------------------------------------------------------------------------
// Prep: W (V,C) f32 -> Wb (VP,C) bf16, zero-padded rows. 2 MB, L2-resident,
// row-major, so each lane's WMMA B-fragment is two 16-byte runs with a 16-byte gap.
// ---------------------------------------------------------------------------
__global__ __launch_bounds__(256) void prep_w(const float* __restrict__ W,
                                              __bf16* __restrict__ Wb) {
  int idx = (blockIdx.x * 256 + threadIdx.x) * 8;   // 2048*512/8 = 512 blocks
  int v = idx >> 9;
  int c = idx & (C_ - 1);
  v8bf o;
  if (v < V_) {
    const float4* p = reinterpret_cast<const float4*>(W + (size_t)v * C_ + c);
    float4 x = p[0], y = p[1];
    o[0]=(__bf16)x.x; o[1]=(__bf16)x.y; o[2]=(__bf16)x.z; o[3]=(__bf16)x.w;
    o[4]=(__bf16)y.x; o[5]=(__bf16)y.y; o[6]=(__bf16)y.z; o[7]=(__bf16)y.w;
  } else {
    for (int j = 0; j < 8; ++j) o[j] = (__bf16)0.0f;
  }
  *reinterpret_cast<v8bf*>(Wb + idx) = o;
}

// ---------------------------------------------------------------------------
// Fused: logits GEMM (bf16 WMMA, B-fragments streamed from L2) + online
// log-softmax + KL(consistency) + gather of ext-label log-probs -> lp_ext.
// Block rows: {16 t's of batch p} ∪ {same t's of batch p+16} so the KL
// partner of row r is row r^16 in the same LDS tile.
// ---------------------------------------------------------------------------
__global__ __launch_bounds__(256) void fused_head(
    const float* __restrict__ enc, const __bf16* __restrict__ Wb,
    const float* __restrict__ bias, const int* __restrict__ lens,
    const int* __restrict__ targets, float* __restrict__ lp_ext,
    float* __restrict__ out)
{
  __shared__ __bf16 As[32][C_];        // 32 KB  encoder rows (full K), bf16
  __shared__ float  Ls[32][128];       // 16 KB  logits tile (f32)
  __shared__ float  red[2][32][8];     //  2 KB  row reductions
  __shared__ float  mrow[32], srow[32], trow[32], tmpm[32];
  __shared__ float  picked[32][208];   // 26 KB  gathered ext-label logits
  __shared__ int    exts[2][S_];       // ext label ids per batch-half
  __shared__ float  klbuf[32];

  const int tid  = threadIdx.x;
  const int t0   = blockIdx.x * 16;
  const int pair = blockIdx.y;          // 0..15
  const int lane = tid & 31;
  const int wv   = tid >> 5;            // wave id 0..7 (wave32)
  const int Nl   = lane & 15;
  const int hl   = lane >> 4;

  // ---- stage A: encoder rows -> LDS bf16 (zero-fill rows past T) ----
  {
    int r  = tid >> 3;                  // 0..31
    int c0 = (tid & 7) * 64;
    int t  = t0 + (r & 15);
    int bg = pair + (r >> 4) * 16;
    if (t < T_) {
      const float4* src = reinterpret_cast<const float4*>(
          enc + ((size_t)(bg * T_ + t)) * C_ + c0);
      for (int j = 0; j < 16; ++j) {
        float4 v = src[j];
        As[r][c0 + 4*j + 0] = (__bf16)v.x;
        As[r][c0 + 4*j + 1] = (__bf16)v.y;
        As[r][c0 + 4*j + 2] = (__bf16)v.z;
        As[r][c0 + 4*j + 3] = (__bf16)v.w;
      }
    } else {
      for (int j = 0; j < 64; ++j) As[r][c0 + j] = (__bf16)0.0f;
    }
  }
  // ext label ids (blank-interleaved) for both batch halves
  for (int idx = tid; idx < 2 * S_; idx += 256) {
    int g  = idx / S_;
    int si = idx - g * S_;
    exts[g][si] = (si & 1) ? targets[(pair + g * 16) * U_ + ((si - 1) >> 1)] : 0;
  }
  if (tid < 32) { mrow[tid] = NEGF; srow[tid] = 0.0f; trow[tid] = 0.0f; }
  __syncthreads();

  // ---- 16 column tiles of 128 labels (VP=2048 covers V=2000) ----
  for (int nt = 0; nt < 16; ++nt) {
    const int n0 = nt * 128;
    v8f acc0 = {0.f,0.f,0.f,0.f,0.f,0.f,0.f,0.f};
    v8f acc1 = {0.f,0.f,0.f,0.f,0.f,0.f,0.f,0.f};
    // this lane's label column, fragment is two 16-byte runs of its Wb row
    const __bf16* wcol = Wb + (size_t)(n0 + wv * 16 + Nl) * C_;

    for (int kc = 0; kc < 16; ++kc) {       // no barriers: A static, B from L2
      const int k0 = kc * 32;
      const v8bf* bp  = reinterpret_cast<const v8bf*>(wcol + k0 + 8 * hl);
      const v8bf* a0p = reinterpret_cast<const v8bf*>(&As[Nl][k0 + 8 * hl]);
      const v8bf* a1p = reinterpret_cast<const v8bf*>(&As[16 + Nl][k0 + 8 * hl]);
      v16bf bfr = cat8(bp[0], bp[2]);
      v16bf a0  = cat8(a0p[0], a0p[2]);     // k: [k0+8h, +8) and [k0+16+8h, +8)
      v16bf a1  = cat8(a1p[0], a1p[2]);
      acc0 = __builtin_amdgcn_wmma_f32_16x16x32_bf16(false, a0, false, bfr,
                                                     (short)0, acc0, false, false);
      acc1 = __builtin_amdgcn_wmma_f32_16x16x32_bf16(false, a1, false, bfr,
                                                     (short)0, acc1, false, false);
    }

    // ---- store accumulators (+bias) to Ls per C-matrix layout ----
    {
      int ncol = wv * 16 + Nl;
      int ng   = n0 + ncol;
      float bv = (ng < V_) ? bias[ng] : 0.0f;
      for (int r = 0; r < 8; ++r) {
        Ls[r + 8*hl][ncol]      = acc0[r] + bv;   // rows 0..15  (batch p)
        Ls[16 + r + 8*hl][ncol] = acc1[r] + bv;   // rows 16..31 (batch p+16)
      }
    }
    __syncthreads();

    // ---- online logsumexp + KL accumulation over this tile ----
    {
      int r  = tid >> 3;
      int lc = tid & 7;
      float lm = NEGF;
      for (int j = 0; j < 16; ++j) {
        int cc = lc*16 + j;
        if (n0 + cc < V_) lm = fmaxf(lm, Ls[r][cc]);
      }
      red[0][r][lc] = lm;
      __syncthreads();
      if (lc == 0) {
        float m = mrow[r];
        for (int j = 0; j < 8; ++j) m = fmaxf(m, red[0][r][j]);
        tmpm[r] = m;
      }
      __syncthreads();
      float mp = tmpm[r];
      float mq = tmpm[r ^ 16];
      float se = 0.f, te = 0.f;
      for (int j = 0; j < 16; ++j) {
        int cc = lc*16 + j;
        if (n0 + cc < V_) {
          float lpv = Ls[r][cc];          // p-row logit
          float lqv = Ls[r ^ 16][cc];     // q-row (paired half) logit
          se += __expf(lpv - mp);
          te += __expf(lqv - mq) * (lqv - lpv);
        }
      }
      red[0][r][lc] = se;
      red[1][r][lc] = te;
      __syncthreads();
      if (lc == 0) {
        float ssum = 0.f, tsum = 0.f;
        for (int j = 0; j < 8; ++j) { ssum += red[0][r][j]; tsum += red[1][r][j]; }
        srow[r] = srow[r] * __expf(mrow[r]      - tmpm[r])      + ssum;
        trow[r] = trow[r] * __expf(mrow[r ^ 16] - tmpm[r ^ 16]) + tsum;
      }
      __syncthreads();
      if (tid < 32) mrow[tid] = tmpm[tid];
    }
    // ---- gather extended-label logits present in this column tile ----
    for (int idx = tid; idx < 32 * S_; idx += 256) {
      int r  = idx / S_;
      int si = idx - r * S_;
      int e  = exts[r >> 4][si];
      if (e >= n0 && e < n0 + 128) picked[r][si] = Ls[r][e - n0];
    }
    __syncthreads();
  }

  // ---- finalize: lse, KL rows (masked), lp_ext writes ----
  if (tid < 32) tmpm[tid] = mrow[tid] + __logf(srow[tid]);   // lse per row
  __syncthreads();
  if (tid < 32) {
    int t  = t0 + (tid & 15);
    int bg = pair + (tid >> 4) * 16;
    float kl = 0.f;
    if (t < T_ && t < lens[bg])
      kl = trow[tid] / srow[tid ^ 16] + (tmpm[tid] - tmpm[tid ^ 16]);
    klbuf[tid] = kl;
  }
  __syncthreads();
  if (tid == 0) {
    float s = 0.f;
    for (int j = 0; j < 32; ++j) s += klbuf[j];
    atomicAdd(&out[1], 0.5f * s);
  }
  for (int idx = tid; idx < 32 * S_; idx += 256) {
    int r  = idx / S_;
    int si = idx - r * S_;
    int t  = t0 + (r & 15);
    if (t < T_) {
      int bg = pair + (r >> 4) * 16;
      lp_ext[((size_t)(bg * T_ + t)) * S_ + si] = picked[r][si] - tmpm[r];
    }
  }
}

// ---------------------------------------------------------------------------
// CTC forward DP: one block per sequence; S=201 states in LDS double buffer.
// ---------------------------------------------------------------------------
__device__ __forceinline__ float lae(float a, float b) {
  float m = fmaxf(a, b);
  if (m <= -1.0e29f) return NEGF;
  float n = fminf(a, b);
  return m + __logf(1.0f + __expf(n - m));
}

__global__ __launch_bounds__(256) void ctc_forward(
    const float* __restrict__ lp, const int* __restrict__ targets,
    const int* __restrict__ lens, const int* __restrict__ tlens,
    float* __restrict__ out)
{
  __shared__ float abuf[2][S_ + 2];
  __shared__ int   extv[S_];
  __shared__ unsigned char skip[S_];
  const int b   = blockIdx.x;
  const int tid = threadIdx.x;

  for (int si = tid; si < S_; si += 256)
    extv[si] = (si & 1) ? targets[b * U_ + ((si - 1) >> 1)] : 0;
  __syncthreads();
  for (int si = tid; si < S_; si += 256) {
    skip[si] = (si >= 2) && (extv[si] != 0) && (extv[si] != extv[si - 2]);
    abuf[0][si + 2] = (si < 2) ? lp[((size_t)(b * T_)) * S_ + si] : NEGF;
    abuf[1][si + 2] = NEGF;
  }
  if (tid < 2) { abuf[0][tid] = NEGF; abuf[1][tid] = NEGF; }
  __syncthreads();

  const int len = lens[b];
  int cur = 0;
  for (int t = 1; t < T_; ++t) {
    const float* lrow = lp + ((size_t)(b * T_ + t)) * S_;
    int nxt = cur ^ 1;
    for (int si = tid; si < S_; si += 256) {
      float a  = abuf[cur][si + 2];
      float a1 = abuf[cur][si + 1];
      float a2 = skip[si] ? abuf[cur][si] : NEGF;
      float v  = lae(lae(a, a1), a2) + lrow[si];
      abuf[nxt][si + 2] = (t < len) ? v : a;   // freeze past input length
    }
    if (t + 1 < T_ && tid < S_)
      __builtin_prefetch(lp + ((size_t)(b * T_ + t + 1)) * S_ + tid, 0, 0);
    __syncthreads();
    cur = nxt;
  }
  if (tid == 0) {
    int end  = 2 * tlens[b];
    float ab = abuf[cur][end + 2];
    float al = abuf[cur][(end > 0 ? end - 1 : 0) + 2];
    atomicAdd(&out[0], -0.5f * lae(ab, al));
  }
}

__global__ void zero_out(float* out) {
  if (threadIdx.x < 2) out[threadIdx.x] = 0.0f;
}

// ---------------------------------------------------------------------------
extern "C" void kernel_launch(void* const* d_in, const int* in_sizes, int n_in,
                              void* d_out, int out_size, void* d_ws, size_t ws_size,
                              hipStream_t stream) {
  const float* enc     = (const float*)d_in[0];   // (32,1000,512)
  const float* W       = (const float*)d_in[1];   // (2000,512)
  const float* bias    = (const float*)d_in[2];   // (2000,)
  const int*   lens    = (const int*)  d_in[3];   // (32,)
  const int*   targets = (const int*)  d_in[4];   // (32,100)
  const int*   tlens   = (const int*)  d_in[5];   // (32,)
  float* out = (float*)d_out;                     // [0]=0.5*ctc, [1]=0.5*cr
  float*  lp = (float*)d_ws;                      // (B,T,S) scratch, 25.7 MB
  __bf16* Wb = (__bf16*)((char*)d_ws + LP_BYTES); // (2048,512) bf16, 2 MB

  zero_out<<<1, 32, 0, stream>>>(out);
  prep_w<<<(VP_ * C_) / (256 * 8), 256, 0, stream>>>(W, Wb);
  dim3 grid((T_ + 15) / 16, B_ / 2);
  fused_head<<<grid, 256, 0, stream>>>(enc, Wb, bias, lens, targets, lp, out);
  ctc_forward<<<B_, 256, 0, stream>>>(lp, targets, lens, tlens, out);
}
